// MMG_8564164788723
// MI455X (gfx1250) — matrix operand order repaired and checked by reading, not compiled, not run on hardware
//
#include <hip/hip_runtime.h>
#include <hip/hip_bf16.h>

// ---------------------------------------------------------------------------
// Problem constants (from reference): B=4, V=256, C1=64, C2=32, H0=H1=128,
// H2=256, OUT=1.
// ---------------------------------------------------------------------------
#define BV    1024          // B*V
#define VN    256           // V
#define CE    32            // C2 (edge attr dim)
#define HD    128           // H0 == H1 (hidden dim)
#define H2D   256           // pair-MLP hidden

typedef _Float16 v16h __attribute__((ext_vector_type(16)));
typedef _Float16 v8h  __attribute__((ext_vector_type(8)));
typedef float    v8f  __attribute__((ext_vector_type(8)));
typedef int      v4i  __attribute__((ext_vector_type(4)));

typedef __attribute__((address_space(1))) v4i* gv4i_p;   // global int4*
typedef __attribute__((address_space(3))) v4i* lv4i_p;   // LDS int4*

union FragH { v16h v; v8h h[2]; };

// ---------------------------------------------------------------------------
// Async global->LDS copy (CDNA5 GLOBAL_LOAD_ASYNC_TO_LDS_B128, ASYNCcnt).
// Probe result (round 2 diagnostic): builtin exists, 4 args, param0 is
// as(1) int4*, so param1 is the as(3) LDS int4* and the trailing (i32, i32)
// are offset / cache-policy immediates.
// ---------------------------------------------------------------------------
#if __has_builtin(__builtin_amdgcn_global_load_async_to_lds_b128)
#define HAVE_ASYNC_LDS 1
#else
#define HAVE_ASYNC_LDS 0
#endif

__device__ __forceinline__ void cp16_to_lds(const _Float16* g, _Float16* l) {
#if HAVE_ASYNC_LDS
    __builtin_amdgcn_global_load_async_to_lds_b128(
        (gv4i_p)(v4i*)(void*)g,
        (lv4i_p)(v4i*)l, 0, 0);
#else
    *(v8h*)l = *(const v8h*)g;
#endif
}

__device__ __forceinline__ void wait_async_lds() {
#if HAVE_ASYNC_LDS
#if __has_builtin(__builtin_amdgcn_s_wait_asynccnt)
    __builtin_amdgcn_s_wait_asynccnt(0);
#else
    asm volatile("s_wait_asynccnt 0" ::: "memory");
#endif
#endif
}

// A-fragment (16x32 f16, M rows): lane ln<16 row M=mbase+ln; half selects the
// K interleave:  lo = K[half*8 .. +7], hi = K[half*8+16 .. +23]   (ISA 7.12.2)
__device__ __forceinline__ v16h load_frag_a(const _Float16* base, int ld,
                                            int mbase, int kbase,
                                            int ln, int half) {
    const _Float16* p = base + (mbase + ln) * ld + kbase + half * 8;
    FragH f;
    f.h[0] = *(const v8h*)(p);
    f.h[1] = *(const v8h*)(p + 16);
    return f.v;
}

// B-fragment (32x16 f16, N cols) loaded from a TRANSPOSED weight tile WT[N][K]:
// lane holds column N=nbase+ln; lo = K[half*8 .. +7], hi = K[half*8+16 .. +23].
__device__ __forceinline__ v16h load_frag_b(const _Float16* base, int ld,
                                            int nbase, int kbase,
                                            int ln, int half) {
    const _Float16* p = base + (nbase + ln) * ld + kbase + half * 8;
    FragH f;
    f.h[0] = *(const v8h*)(p);
    f.h[1] = *(const v8h*)(p + 16);
    return f.v;
}

// ---------------------------------------------------------------------------
// Kernel 0: one-shot weight prep. Transpose + convert to f16:
//   W1eT[n][k] = W1e[k][n]  (CE x HD -> HD x CE)
//   W2T [n][k] = W2 [k][n]  (HD x HD -> HD x HD)
// grid = 2 (layer), block = 256
// ---------------------------------------------------------------------------
__global__ void weight_prep_kernel(const float* __restrict__ W1e_1,
                                   const float* __restrict__ W2_1,
                                   const float* __restrict__ W1e_2,
                                   const float* __restrict__ W2_2,
                                   _Float16* __restrict__ W1eT_1,
                                   _Float16* __restrict__ W2T_1,
                                   _Float16* __restrict__ W1eT_2,
                                   _Float16* __restrict__ W2T_2) {
    const float* W1e = blockIdx.x ? W1e_2 : W1e_1;
    const float* W2  = blockIdx.x ? W2_2  : W2_1;
    _Float16* T1 = blockIdx.x ? W1eT_2 : W1eT_1;
    _Float16* T2 = blockIdx.x ? W2T_2  : W2T_1;
    for (int s = threadIdx.x; s < CE * HD; s += 256) {
        const int k = s >> 7, n = s & 127;
        T1[n * CE + k] = (_Float16)W1e[s];
    }
    for (int s = threadIdx.x; s < HD * HD; s += 256) {
        const int k = s >> 7, n = s & 127;
        T2[n * HD + k] = (_Float16)W2[s];
    }
}

// ---------------------------------------------------------------------------
// Kernel 1: per-node projections for an EdgeConv layer.
//   P[m,n] = b1[n] + sum_k x[m,k] * (W1[k,n] - W1[C+k,n])     (center term)
//   Q[m,n] =         sum_k x[m,k] *  W1[C+k,n]                (neighbor term)
// grid = BV, block = HD
// ---------------------------------------------------------------------------
__global__ void node_proj_kernel(const float* __restrict__ X,
                                 const float* __restrict__ W1,
                                 const float* __restrict__ b1,
                                 float* __restrict__ P,
                                 float* __restrict__ Q, int C) {
    extern __shared__ char smem[];
    float* sx = (float*)smem;                       // C floats
    const int m = blockIdx.x, n = threadIdx.x;
    for (int k = n; k < C; k += HD) sx[k] = X[(long)m * C + k];
    __syncthreads();
    float pa = b1[n], qa = 0.f;
    for (int k = 0; k < C; ++k) {
        const float x  = sx[k];
        const float wa = W1[k * HD + n];
        const float wb = W1[(C + k) * HD + n];
        pa = fmaf(x, wa - wb, pa);
        qa = fmaf(x, wb, qa);
    }
    P[(long)m * HD + n] = pa;
    Q[(long)m * HD + n] = qa;
}

// ---------------------------------------------------------------------------
// Kernel 2: fused EdgeConvE over all neighbors of one (b,i).
//   h1[j,:] = relu( e_ij @ W1e + P_i + Q_j )         (WMMA, K=32)
//   h2[j,:] = relu( h1 @ W2 + b2 )                   (WMMA, K=128)
//   out[i,:] = max_{j: adj=1} h2[j,:]  (0 if isolated; h2>=0 so uint-max works)
// grid = BV (one WG per (b,i)), block = 256 (8 wave32)
// ---------------------------------------------------------------------------
#define LDE (CE + 8)              // 40 f16  (16B-aligned rows)
#define LDH (HD + 8)              // 136 f16 (16B-aligned rows)
#define SZ_E   (VN * LDE * 2)     // 20480
#define SZ_W1  (HD * LDE * 2)     // 10240
#define SZ_W2  (HD * LDH * 2)     // 34816
#define SZ_H1  (VN * LDH * 2)     // 69632
#define EDGE_SMEM (SZ_E + SZ_W1 + SZ_W2 + SZ_H1 + 3 * HD * 4 + VN * 4) // 137728

__global__ void __launch_bounds__(256, 2)
edgeconv_wmma_kernel(const float*    __restrict__ Eattr,  // (B,V,V,CE) f32
                     const int*      __restrict__ adj,    // (B,V,V)
                     const float*    __restrict__ P,      // (BV,HD)  bias folded
                     const float*    __restrict__ Q,      // (BV,HD)
                     const _Float16* __restrict__ W1eT,   // (HD,CE) f16, pre-T
                     const _Float16* __restrict__ W2T,    // (HD,HD) f16, pre-T
                     const float*    __restrict__ b2,     // (HD)
                     float* __restrict__ Xout) {          // (BV,HD)
    extern __shared__ char smem[];
    _Float16* sE   = (_Float16*)smem;                          // [VN][LDE]
    _Float16* sW1  = (_Float16*)(smem + SZ_E);                 // [HD][LDE]
    _Float16* sW2  = (_Float16*)(smem + SZ_E + SZ_W1);         // [HD][LDH]
    _Float16* sH1  = (_Float16*)(smem + SZ_E + SZ_W1 + SZ_W2); // [VN][LDH]
    float*    sP   = (float*)(smem + SZ_E + SZ_W1 + SZ_W2 + SZ_H1);
    float*    sB2  = sP + HD;
    unsigned* sOut = (unsigned*)(sB2 + HD);
    int*      sAdj = (int*)(sOut + HD);

    const int bi = blockIdx.x;            // b*V + i
    const int b  = bi >> 8;
    const int t  = threadIdx.x;
    const float* Qb = Q + ((long)b << 8) * HD;   // rows Q[b*V + j]

    // ---- async-stage pre-converted f16 weights into padded LDS ----
    #pragma unroll
    for (int s = t; s < (HD * CE) / 8; s += 256) {         // 512 x 16B
        const int n = s >> 2, koff = (s & 3) * 8;
        cp16_to_lds(W1eT + s * 8, sW1 + n * LDE + koff);
    }
    #pragma unroll
    for (int s = t; s < (HD * HD) / 8; s += 256) {         // 2048 x 16B
        const int n = s >> 4, koff = (s & 15) * 8;
        cp16_to_lds(W2T + s * 8, sW2 + n * LDH + koff);
    }

    // ---- stage E tile (f32 -> f16) and small vectors (overlaps async) ----
    const long ebase = (long)bi * (VN * CE);
    for (int s = t; s < VN * CE; s += 256) {
        const int j = s >> 5, k = s & 31;
        sE[j * LDE + k] = (_Float16)Eattr[ebase + s];
    }
    if (t < HD) {
        sP[t]   = P[(long)bi * HD + t];
        sB2[t]  = b2[t];
        sOut[t] = 0u;                            // == bits of 0.0f
    }
    sAdj[t] = adj[(long)bi * VN + t];
    __builtin_prefetch(Qb + (long)t * HD, 0, 1); // global_prefetch_b8 on Q rows
    wait_async_lds();
    __syncthreads();

    const int wave = t >> 5;
    const int lane = t & 31;
    const int ln   = lane & 15;
    const int half = lane >> 4;

    // ---- G1: h1 = relu(E @ W1e + P_i + Q_j), K = 32 (one WMMA per tile) ----
    #pragma unroll
    for (int mt = 0; mt < 2; ++mt) {
        const int mbase = wave * 32 + mt * 16;
        const v16h a = load_frag_a(sE, LDE, mbase, 0, ln, half);
        #pragma unroll
        for (int nt = 0; nt < 8; ++nt) {
            const int nbase = nt * 16;
            const int n = nbase + ln;
            v8f c;
            #pragma unroll
            for (int r = 0; r < 8; ++r) {
                const int j = mbase + half * 8 + r;
                c[r] = sP[n] + Qb[(long)j * HD + n];
            }
            const v16h bf = load_frag_b(sW1, LDE, nbase, 0, ln, half);
            c = __builtin_amdgcn_wmma_f32_16x16x32_f16(false, a, false, bf,
                                                       (short)0, c, false, false);
            #pragma unroll
            for (int r = 0; r < 8; ++r) {
                const int j = mbase + half * 8 + r;
                const float v = c[r] > 0.f ? c[r] : 0.f;
                sH1[j * LDH + n] = (_Float16)v;
            }
        }
    }
    __syncthreads();

    // ---- G2: h2 = relu(h1 @ W2 + b2), K = 128; masked max-reduce over j ----
    #pragma unroll
    for (int mt = 0; mt < 2; ++mt) {
        const int mbase = wave * 32 + mt * 16;
        #pragma unroll
        for (int nt = 0; nt < 8; ++nt) {
            const int nbase = nt * 16;
            const int n = nbase + ln;
            v8f c;
            #pragma unroll
            for (int r = 0; r < 8; ++r) c[r] = sB2[n];
            #pragma unroll
            for (int kt = 0; kt < 4; ++kt) {
                const v16h a  = load_frag_a(sH1, LDH, mbase, kt * 32, ln, half);
                const v16h bf = load_frag_b(sW2, LDH, nbase, kt * 32, ln, half);
                c = __builtin_amdgcn_wmma_f32_16x16x32_f16(false, a, false, bf,
                                                           (short)0, c, false, false);
            }
            float vmax = 0.f;
            #pragma unroll
            for (int r = 0; r < 8; ++r) {
                const int j = mbase + half * 8 + r;
                const float v = c[r] > 0.f ? c[r] : 0.f;
                if (sAdj[j] != 0) vmax = fmaxf(vmax, v);
            }
            atomicMax(&sOut[n], __float_as_uint(vmax));   // ds_max_u32
        }
    }
    __syncthreads();
    if (t < HD) Xout[(long)bi * HD + t] = __uint_as_float(sOut[t]);
}

// ---------------------------------------------------------------------------
// Kernel 3: pair-MLP per-node projections.
//   R[m,n] = sum_k X[m,k] * W3[k,n]        (neighbor/src half)
//   S[m,n] = sum_k X[m,k] * W3[128+k,n]    (center/tgt half)
// grid = BV, block = 256
// ---------------------------------------------------------------------------
__global__ void pair_proj_kernel(const float* __restrict__ X,
                                 const float* __restrict__ W3,
                                 float* __restrict__ R,
                                 float* __restrict__ S) {
    extern __shared__ char smem[];
    float* sx = (float*)smem;                        // 128 floats
    const int m = blockIdx.x, n = threadIdx.x;
    if (n < HD) sx[n] = X[(long)m * HD + n];
    __syncthreads();
    float r = 0.f, s = 0.f;
    for (int k = 0; k < HD; ++k) {
        const float x = sx[k];
        r = fmaf(x, W3[k * H2D + n], r);
        s = fmaf(x, W3[(HD + k) * H2D + n], s);
    }
    R[(long)m * H2D + n] = r;
    S[(long)m * H2D + n] = s;
}

// ---------------------------------------------------------------------------
// Kernel 4: pair scoring (nonlinear part only, cannot be factored).
//   out[b,i,j] = sigmoid( sum_h relu(R[b,j,h] + S[b,i,h] + b3[h]) * outW[h] + outb )
// grid = BV (one WG per (b,i)), block = 256 (j)
// ---------------------------------------------------------------------------
__global__ void pair_score_kernel(const float* __restrict__ R,
                                  const float* __restrict__ S,
                                  const float* __restrict__ b3,
                                  const float* __restrict__ outW,
                                  const float* __restrict__ outb,
                                  float* __restrict__ out) {
    __shared__ float sT[H2D];
    __shared__ float sW[H2D];
    const int bi = blockIdx.x;
    const int b  = bi >> 8;
    const int j  = threadIdx.x;
    sT[j] = S[(long)bi * H2D + j] + b3[j];
    sW[j] = outW[j];
    __syncthreads();
    const float* Rj = R + ((long)((b << 8) | j)) * H2D;
    float acc = outb[0];
    for (int h = 0; h < H2D; ++h) {
        const float v = Rj[h] + sT[h];
        acc = fmaf(fmaxf(v, 0.f), sW[h], acc);
    }
    out[(long)bi * VN + j] = 1.f / (1.f + __expf(-acc));
}

// ---------------------------------------------------------------------------
extern "C" void kernel_launch(void* const* d_in, const int* in_sizes, int n_in,
                              void* d_out, int out_size, void* d_ws, size_t ws_size,
                              hipStream_t stream) {
    const int*   adj   = (const int*)  d_in[0];
    const float* xfeat = (const float*)d_in[1];
    const float* eattr = (const float*)d_in[2];
    const float* e1W1  = (const float*)d_in[3];
    const float* e1b1  = (const float*)d_in[4];
    const float* e1W2  = (const float*)d_in[5];
    const float* e1b2  = (const float*)d_in[6];
    const float* e2W1  = (const float*)d_in[7];
    const float* e2b1  = (const float*)d_in[8];
    const float* e2W2  = (const float*)d_in[9];
    const float* e2b2  = (const float*)d_in[10];
    const float* h3W   = (const float*)d_in[11];
    const float* h3b   = (const float*)d_in[12];
    const float* outW  = (const float*)d_in[13];
    const float* outb  = (const float*)d_in[14];
    float* out = (float*)d_out;

    // workspace carve: 6*BV*HD + 2*BV*H2D f32, then 2*(HD*CE + HD*HD) f16
    float* P1 = (float*)d_ws;
    float* Q1 = P1 + BV * HD;
    float* X1 = Q1 + BV * HD;
    float* P2 = X1 + BV * HD;
    float* Q2 = P2 + BV * HD;
    float* X2 = Q2 + BV * HD;
    float* R  = X2 + BV * HD;
    float* S  = R  + BV * H2D;
    _Float16* w1t1 = (_Float16*)(S + BV * H2D);
    _Float16* w2t1 = w1t1 + HD * CE;
    _Float16* w1t2 = w2t1 + HD * HD;
    _Float16* w2t2 = w1t2 + HD * CE;

    // One-shot weight transpose/convert (W1e = trailing CE rows of ecX_W1).
    weight_prep_kernel<<<2, 256, 0, stream>>>(
        e1W1 + 2 * 64 * HD, e1W2, e2W1 + 2 * 128 * HD, e2W2,
        w1t1, w2t1, w1t2, w2t2);

    // Layer 1 (C=64).
    node_proj_kernel<<<BV, HD, 64 * sizeof(float), stream>>>(
        xfeat, e1W1, e1b1, P1, Q1, 64);
    edgeconv_wmma_kernel<<<BV, 256, EDGE_SMEM, stream>>>(
        eattr, adj, P1, Q1, w1t1, w2t1, e1b2, X1);

    // Layer 2 (C=128).
    node_proj_kernel<<<BV, HD, 128 * sizeof(float), stream>>>(
        X1, e2W1, e2b1, P2, Q2, 128);
    edgeconv_wmma_kernel<<<BV, 256, EDGE_SMEM, stream>>>(
        eattr, adj, P2, Q2, w1t2, w2t2, e2b2, X2);

    // Pair scorer.
    pair_proj_kernel<<<BV, 256, 128 * sizeof(float), stream>>>(X2, h3W, R, S);
    pair_score_kernel<<<BV, 256, 0, stream>>>(R, S, h3b, outW, outb, out);
}
